// WorldModel_5961414607295
// MI455X (gfx1250) — matrix operand order repaired and checked by reading, not compiled, not run on hardware
//
#include <hip/hip_runtime.h>
#include <hip/hip_bf16.h>
#include <cstdint>
#include <cstddef>

// ---------------- types / device helpers ----------------
typedef __attribute__((ext_vector_type(16))) _Float16 v16h;
typedef __attribute__((ext_vector_type(8)))  _Float16 v8h;
typedef __attribute__((ext_vector_type(4)))  _Float16 v4h;
typedef __attribute__((ext_vector_type(8)))  float    v8f;
typedef __attribute__((ext_vector_type(4)))  int      v4i;

#define DI __device__ __forceinline__

DI float d_sigmoid(float x) { return 1.f / (1.f + __expf(-x)); }
DI float d_act(float v, int act) {
    switch (act) {
        case 1: return v * d_sigmoid(v);                      // silu
        case 2: return tanhf(v);                              // tanh
        case 3: return v > 0.f ? v : (__expf(v) - 1.f);       // elu
        default: return v;
    }
}

// block-wide sum (blockDim.x == 256) then atomic accumulate
DI void block_sum_atomic(float v, float* target) {
    __shared__ float red[256];
    __syncthreads();
    red[threadIdx.x] = v;
    __syncthreads();
    for (int s = 128; s > 0; s >>= 1) {
        if ((int)threadIdx.x < s) red[threadIdx.x] += red[threadIdx.x + s];
        __syncthreads();
    }
    if (threadIdx.x == 0) atomicAdd(target, red[0]);
}

#define GS_LOOP(i, n) for (size_t i = (size_t)blockIdx.x * blockDim.x + threadIdx.x; \
                           i < (n); i += (size_t)gridDim.x * blockDim.x)

// ---------------- async global->LDS copy (CDNA5), with portable fallback ----------------
#if __has_builtin(__builtin_amdgcn_global_load_async_to_lds_b128)
#define HAVE_ASYNC_LDS 1
#else
#define HAVE_ASYNC_LDS 0
#endif

#if HAVE_ASYNC_LDS
typedef __attribute__((address_space(1))) v4i gv4i;   // global-memory 16B payload
typedef __attribute__((address_space(3))) v4i lv4i;   // LDS 16B payload
#endif

// copy 32 bytes (16 halves) global -> LDS for this lane
DI void b_tile_copy(const _Float16* gsrc, _Float16* ldst) {
#if HAVE_ASYNC_LDS
    __builtin_amdgcn_global_load_async_to_lds_b128((gv4i*)gsrc, (lv4i*)ldst, 0, 0);
    __builtin_amdgcn_global_load_async_to_lds_b128((gv4i*)(gsrc + 8), (lv4i*)(ldst + 8), 0, 0);
#else
    *(v8h*)ldst       = *(const v8h*)gsrc;
    *(v8h*)(ldst + 8) = *(const v8h*)(gsrc + 8);
#endif
}

DI void b_copy_wait() {
#if HAVE_ASYNC_LDS
#if __has_builtin(__builtin_amdgcn_s_wait_asynccnt)
    __builtin_amdgcn_s_wait_asynccnt(0);
#else
    asm volatile("s_wait_asynccnt 0x0" ::: "memory");
#endif
#endif
}

// ================= WMMA GEMM =================
// C = act(A(f32, MxK row-major) * Bsw(f16, WMMA-swizzled KxN) + bias)
// B swizzle: tile (ntA, kt) base = (ntA*KT + kt)*512 halves; within tile lane l
// owns 16 contiguous halves j=0..15 holding B[kt*32 + 16*(l>>4) + j][ntA*16 + (l&15)].
// Per K-step each wave async-copies ONE 1KB B tile into LDS (block shares all 4),
// quartering redundant global B traffic; A is staged wave-privately f32->f16.
// P==0: plain row-major store; P>0: NCHW scatter (row==(image,pixel), col==channel).
#define BM 64
#define BN 64
#define BK 32

DI v8f wmma_step(const v16h& a, const v16h& b, const v8f& c) {
    return __builtin_amdgcn_wmma_f32_16x16x32_f16(false, a, false, b, (short)0, c,
                                                  false, false);
}

template <bool FULLN>
DI void gemm_kloop(const float* __restrict__ A, const _Float16* __restrict__ Bsw,
                   _Float16* As, _Float16* Bls, int M, int K, int KT, int NT,
                   int rowBase, int bx, int wave, int lane, int lh, int hi, bool vec4,
                   v8f& acc0, v8f& acc1, v8f& acc2, v8f& acc3)
{
    const bool myTile = FULLN || (bx * 4 + wave < NT);
    for (int kt = 0; kt < KT; ++kt) {
        const int k0 = kt << 5;

        __syncthreads();   // previous iteration's Bls readers are done

        // ---- kick off this wave's 1KB B-tile copy into LDS (ASYNCcnt path) ----
        if (myTile) {
            const _Float16* gsrc =
                Bsw + (((size_t)(bx * 4 + wave) * KT + kt) << 9) + ((size_t)lane << 4);
            b_tile_copy(gsrc, Bls + wave * 512 + lane * 16);
        }

        // ---- stage this wave's own 16x32 A sub-tile (f32 -> f16) meanwhile ----
        #pragma unroll
        for (int q = 0; q < 4; ++q) {
            const int e4 = q * 32 + lane;                 // 128 quads per wave region
            const int r  = wave * 16 + (e4 >> 3);
            const int c4 = (e4 & 7) * 4;
            const int gr = rowBase + r;
            v4h hv;
            if (vec4 && gr < M && (k0 + c4 + 3) < K) {
                const float4 f = *(const float4*)(A + (size_t)gr * K + k0 + c4);
                hv[0] = (_Float16)f.x; hv[1] = (_Float16)f.y;
                hv[2] = (_Float16)f.z; hv[3] = (_Float16)f.w;
            } else {
                #pragma unroll
                for (int u = 0; u < 4; ++u) {
                    const int gc = k0 + c4 + u;
                    hv[u] = (gr < M && gc < K) ? (_Float16)A[(size_t)gr * K + gc]
                                               : (_Float16)0.f;
                }
            }
            *(v4h*)(As + r * BK + c4) = hv;
        }
        // speculative prefetch of next A tile (global_prefetch_b8 path)
        if (k0 + BK < K) {
            const int pr = rowBase + wave * 16 + lh;
            if (pr < M) __builtin_prefetch(&A[(size_t)pr * K + k0 + BK], 0, 0);
        }

        // ---- A fragment: two contiguous 16B LDS loads + shuffle ----
        const v8h* arow = (const v8h*)(As + (wave * 16 + lh) * BK);
        const v8h a0 = arow[2 * hi];      // K = 16*hi .. 16*hi+7
        const v8h a1 = arow[2 * hi + 1];  // K = 16*hi+8 .. 16*hi+15
        const v16h af = __builtin_shufflevector(a0, a1,
            0, 1, 2, 3, 4, 5, 6, 7, 8, 9, 10, 11, 12, 13, 14, 15);

        b_copy_wait();     // own async copies complete
        __syncthreads();   // all four tiles visible block-wide

        // ---- B fragments from LDS (2x ds_load_b128 each) + 4 WMMAs ----
        const int lo = lane * 16;
        if (FULLN) {
            const v16h bf0 = *(const v16h*)(Bls + 0 * 512 + lo);
            const v16h bf1 = *(const v16h*)(Bls + 1 * 512 + lo);
            const v16h bf2 = *(const v16h*)(Bls + 2 * 512 + lo);
            const v16h bf3 = *(const v16h*)(Bls + 3 * 512 + lo);
            acc0 = wmma_step(af, bf0, acc0);
            acc1 = wmma_step(af, bf1, acc1);
            acc2 = wmma_step(af, bf2, acc2);
            acc3 = wmma_step(af, bf3, acc3);
        } else {
            if (bx * 4 + 0 < NT) { const v16h b = *(const v16h*)(Bls + 0 * 512 + lo);
                                   acc0 = wmma_step(af, b, acc0); }
            if (bx * 4 + 1 < NT) { const v16h b = *(const v16h*)(Bls + 1 * 512 + lo);
                                   acc1 = wmma_step(af, b, acc1); }
            if (bx * 4 + 2 < NT) { const v16h b = *(const v16h*)(Bls + 2 * 512 + lo);
                                   acc2 = wmma_step(af, b, acc2); }
            if (bx * 4 + 3 < NT) { const v16h b = *(const v16h*)(Bls + 3 * 512 + lo);
                                   acc3 = wmma_step(af, b, acc3); }
        }
    }
}

DI void gemm_store_tile(const v8f& accv, int nt,
                        const float* __restrict__ bias, float* __restrict__ C,
                        int M, int N, int mOff, int biasDiv, int act, int P,
                        int rowBase, int colBase, int wave, int lh, int hi)
{
    const int nn = colBase + nt * 16 + lh;
    #pragma unroll
    for (int r = 0; r < 8; ++r) {
        const int mm = rowBase + wave * 16 + r + 8 * hi;
        if (mm < M && nn < N) {
            float v = accv[r];
            if (bias) v += bias[nn / biasDiv];
            v = d_act(v, act);
            const long gm = (long)mm + mOff;
            size_t idx;
            if (P > 0) {
                long nimg = gm / P, pix = gm % P;
                idx = ((size_t)nimg * N + nn) * (size_t)P + pix;
            } else {
                idx = (size_t)gm * N + nn;
            }
            C[idx] = v;
        }
    }
}

__global__ __launch_bounds__(128) void k_wmma_gemm(
    const float* __restrict__ A, const _Float16* __restrict__ Bsw,
    const float* __restrict__ bias, float* __restrict__ C,
    int M, int N, int K, int mOff, int biasDiv, int act, int P)
{
    __shared__ alignas(32) _Float16 As[BM * BK];   // wave-private 16x32 sub-tiles
    __shared__ alignas(32) _Float16 Bls[4 * 512];  // block-shared B tiles (4KB)

    const int tid  = threadIdx.x;
    const int wave = tid >> 5;
    const int lane = tid & 31;
    const int lh   = lane & 15;
    const int hi   = lane >> 4;

    const int bx      = blockIdx.x;
    const int rowBase = blockIdx.y * BM;
    const int colBase = bx * BN;
    const int KT = (K + 31) >> 5;
    const int NT = (N + 15) >> 4;
    const bool vec4 = ((K & 3) == 0);

    v8f acc0 = {}, acc1 = {}, acc2 = {}, acc3 = {};

    if (bx * 4 + 3 < NT) {
        gemm_kloop<true>(A, Bsw, As, Bls, M, K, KT, NT, rowBase, bx,
                         wave, lane, lh, hi, vec4, acc0, acc1, acc2, acc3);
    } else {
        gemm_kloop<false>(A, Bsw, As, Bls, M, K, KT, NT, rowBase, bx,
                          wave, lane, lh, hi, vec4, acc0, acc1, acc2, acc3);
    }

    gemm_store_tile(acc0, 0, bias, C, M, N, mOff, biasDiv, act, P, rowBase, colBase, wave, lh, hi);
    gemm_store_tile(acc1, 1, bias, C, M, N, mOff, biasDiv, act, P, rowBase, colBase, wave, lh, hi);
    gemm_store_tile(acc2, 2, bias, C, M, N, mOff, biasDiv, act, P, rowBase, colBase, wave, lh, hi);
    gemm_store_tile(acc3, 3, bias, C, M, N, mOff, biasDiv, act, P, rowBase, colBase, wave, lh, hi);
}

// ---------------- B-swizzle pack kernels ----------------
// decode swizzled element e -> (krow, ncol)
DI void swz_decode(size_t e, int KT, int& krow, int& ncol) {
    const size_t tile = e >> 9;
    const int rem  = (int)(e & 511);
    const int lane = rem >> 4, j = rem & 15;
    const int ntA  = (int)(tile / KT), kt = (int)(tile % KT);
    krow = kt * 32 + (lane >> 4) * 16 + j;
    ncol = ntA * 16 + (lane & 15);
}

// conv OIHW (oc,ic,kh,kw) -> swizzled [K=ic*KHW][N=oc]
__global__ void k_pack_conv_w(const float* w, _Float16* o, int OC, int IC, int KHW, int KT) {
    const int NT = (OC + 15) >> 4;
    const int K  = IC * KHW;
    size_t n = (size_t)KT * NT * 512;
    GS_LOOP(e, n) {
        int krow, ncol; swz_decode(e, KT, krow, ncol);
        float v = 0.f;
        if (krow < K && ncol < OC) {
            int ic = krow / KHW, r = krow % KHW;
            v = w[((size_t)ncol * IC + ic) * KHW + r];
        }
        o[e] = (_Float16)v;
    }
}
// deconv IOHW (ic,oc,kh,kw) -> swizzled [K=ic*KHW][N=oc]
__global__ void k_pack_deconv_w(const float* w, _Float16* o, int IC, int OC, int KHW, int KT) {
    const int NT = (OC + 15) >> 4;
    const int K  = IC * KHW;
    size_t n = (size_t)KT * NT * 512;
    GS_LOOP(e, n) {
        int krow, ncol; swz_decode(e, KT, krow, ncol);
        float v = 0.f;
        if (krow < K && ncol < OC) {
            int ic = krow / KHW, r = krow % KHW;
            v = w[((size_t)ic * OC + ncol) * KHW + r];
        }
        o[e] = (_Float16)v;
    }
}
// linear W(OUT,IN) -> swizzled transposed [K=IN][N=OUT]  (GEMM computes x @ W^T)
__global__ void k_pack_linear_w(const float* w, _Float16* o, int OUT, int IN, int KT) {
    const int NT = (OUT + 15) >> 4;
    size_t n = (size_t)KT * NT * 512;
    GS_LOOP(e, n) {
        int krow, ncol; swz_decode(e, KT, krow, ncol);
        float v = (krow < IN && ncol < OUT) ? w[(size_t)ncol * IN + krow] : 0.f;
        o[e] = (_Float16)v;
    }
}
// plain row-major [K][N] f32 -> swizzled (used for Wd1 reshape)
__global__ void k_pack_rowmajor_w(const float* w, _Float16* o, int K, int N, int KT) {
    const int NT = (N + 15) >> 4;
    size_t n = (size_t)KT * NT * 512;
    GS_LOOP(e, n) {
        int krow, ncol; swz_decode(e, KT, krow, ncol);
        float v = (krow < K && ncol < N) ? w[(size_t)krow * N + ncol] : 0.f;
        o[e] = (_Float16)v;
    }
}

// ---------------- misc utility ----------------
__global__ void k_zero(float* p, size_t n) { GS_LOOP(i, n) p[i] = 0.f; }

// ---------------- im2col (conv / conv_transpose), f32 patches ----------------
__global__ void k_im2col_conv(const float* __restrict__ in, float* __restrict__ patch,
    int mOff, int Mc, int IC, int IH, int IW, int OH, int OW, int KH, int KW, int S, int Pd)
{
    const int K = IC * KH * KW;
    size_t n = (size_t)Mc * K;
    GS_LOOP(e, n) {
        int m = (int)(e / K), k = (int)(e % K);
        int gm = m + mOff;
        int img = gm / (OH * OW), pix = gm % (OH * OW);
        int oy = pix / OW, ox = pix % OW;
        int ic = k / (KH * KW), r = k % (KH * KW);
        int kh = r / KW, kw = r % KW;
        int iy = oy * S - Pd + kh, ix = ox * S - Pd + kw;
        float v = 0.f;
        if (iy >= 0 && iy < IH && ix >= 0 && ix < IW)
            v = in[(((size_t)img * IC + ic) * IH + iy) * IW + ix];
        patch[e] = v;
    }
}

__global__ void k_im2col_deconv(const float* __restrict__ in, float* __restrict__ patch,
    int mOff, int Mc, int IC, int IH, int IW, int OH, int OW, int KH, int KW, int S, int Pd)
{
    // transposed conv: oy = iy*S - Pd + kh  =>  iy = (oy + Pd - kh)/S when divisible
    const int K = IC * KH * KW;
    size_t n = (size_t)Mc * K;
    GS_LOOP(e, n) {
        int m = (int)(e / K), k = (int)(e % K);
        int gm = m + mOff;
        int img = gm / (OH * OW), pix = gm % (OH * OW);
        int oy = pix / OW, ox = pix % OW;
        int ic = k / (KH * KW), r = k % (KH * KW);
        int kh = r / KW, kw = r % KW;
        int iyn = oy + Pd - kh, ixn = ox + Pd - kw;
        float v = 0.f;
        if (iyn >= 0 && ixn >= 0 && (iyn % S) == 0 && (ixn % S) == 0) {
            int iy = iyn / S, ix = ixn / S;
            if (iy < IH && ix < IW)
                v = in[(((size_t)img * IC + ic) * IH + iy) * IW + ix];
        }
        patch[e] = v;
    }
}

// ---------------- model glue kernels ----------------
__global__ void k_reparam(const float* mu, const float* lv, const float* eps, float* z, size_t n) {
    GS_LOOP(i, n) z[i] = mu[i] + __expf(0.5f * lv[i]) * eps[i];
}

// x = concat(mu[:,t], actions[:,t]) -> (16,134)
__global__ void k_concat_scan(const float* __restrict__ muf, const float* __restrict__ act,
                              float* __restrict__ x, int t) {
    size_t n = (size_t)16 * 134;
    GS_LOOP(e, n) {
        int b = (int)(e / 134), j = (int)(e % 134);
        x[e] = (j < 128) ? muf[((size_t)b * 16 + t) * 128 + j]
                         : act[((size_t)b * 15 + t) * 6 + (j - 128)];
    }
}

// rollout concat: rows m = s*16+b, s in [0,13)
__global__ void k_concat_roll(const float* __restrict__ zr, const float* __restrict__ act,
                              float* __restrict__ x, int k) {
    size_t n = (size_t)208 * 134;
    GS_LOOP(e, n) {
        int m = (int)(e / 134), j = (int)(e % 134);
        int s = m / 16, b = m % 16;
        x[e] = (j < 128) ? zr[(size_t)m * 128 + j]
                         : act[((size_t)b * 15 + (k + s)) * 6 + (j - 128)];
    }
}

__global__ void k_gru_combine(const float* gi, const float* gh, const float* h, float* ho, int M) {
    size_t n = (size_t)M * 256;
    GS_LOOP(e, n) {
        int m = (int)(e / 256), d = (int)(e % 256);
        const float* gim = gi + (size_t)m * 768;
        const float* ghm = gh + (size_t)m * 768;
        float r  = d_sigmoid(gim[d] + ghm[d]);
        float u  = d_sigmoid(gim[256 + d] + ghm[256 + d]);
        float nn = tanhf(gim[512 + d] + r * ghm[512 + d]);
        ho[e] = (1.f - u) * nn + u * h[e];
    }
}

__global__ void k_prior_post(const float* p, float* pm, float* plv, int M) {
    size_t n = (size_t)M * 128;
    GS_LOOP(e, n) {
        int m = (int)(e / 128), f = (int)(e % 128);
        pm[e] = p[(size_t)m * 256 + f];
        float raw = p[(size_t)m * 256 + 128 + f];
        float sp  = raw > 20.f ? raw : log1pf(__expf(raw));
        plv[e] = 2.f * logf(sp + 0.1f);
    }
}

// ---------------- reduction kernels (blockDim 256) ----------------
__global__ void k_kl0(const float* muf, const float* lvf, float* acc) {
    float s = 0.f;
    size_t n = (size_t)16 * 128;
    GS_LOOP(e, n) {
        int b = (int)(e / 128), f = (int)(e % 128);
        float m0 = muf[((size_t)b * 16) * 128 + f];
        float l0 = lvf[((size_t)b * 16) * 128 + f];
        s += 0.5f * (m0 * m0 + __expf(l0) - 1.f - l0);
    }
    block_sum_atomic(s, acc + 1);
}

__global__ void k_drift(const float* muf, float* acc) {
    float s = 0.f;
    size_t n = (size_t)16 * 15 * 128;
    GS_LOOP(e, n) {
        int f = (int)(e % 128); int bt = (int)(e / 128);
        int b = bt / 15, t = bt % 15;
        float a = muf[((size_t)b * 16 + t + 1) * 128 + f];
        float c = muf[((size_t)b * 16 + t) * 128 + f];
        s += fabsf(a - c);
    }
    block_sum_atomic(s, acc + 4);
}

__global__ void k_kl_err_step(const float* pm, const float* plv,
                              const float* muf, const float* lvf, int t, float* acc) {
    float skl = 0.f, serr = 0.f;
    size_t n = (size_t)16 * 128;
    GS_LOOP(e, n) {
        int b = (int)(e / 128), f = (int)(e % 128);
        float mq = muf[((size_t)b * 16 + t + 1) * 128 + f];
        float lq = lvf[((size_t)b * 16 + t + 1) * 128 + f];
        float mp = pm[e], lp = plv[e];
        float d = mq - mp;
        skl  += 0.5f * (lp - lq + (__expf(lq) + d * d) * __expf(-lp) - 1.f);
        serr += d * d;
    }
    block_sum_atomic(skl,  acc + 1);
    block_sum_atomic(serr, acc + 2);
}

__global__ void k_roll_err(const float* pm, const float* muf, int k, float* acc) {
    float s = 0.f;
    size_t n = (size_t)208 * 128;
    GS_LOOP(e, n) {
        int m = (int)(e / 128), f = (int)(e % 128);
        int ss = m / 16, b = m % 16;
        float d = pm[e] - muf[((size_t)b * 16 + (k + 1 + ss)) * 128 + f];
        s += d * d;
    }
    block_sum_atomic(s, acc + 3);
}

__global__ void k_recloss(const float* xr, const float* img, float* acc, size_t n) {
    float s = 0.f;
    GS_LOOP(e, n) { float d = xr[e] - img[e]; s += d * d; }
    block_sum_atomic(s, acc + 0);
}

__global__ void k_roll_init(const float* hs, const float* muf, float* hr, float* zr) {
    size_t n = (size_t)208 * 256;
    GS_LOOP(e, n) {
        int m = (int)(e / 256), d = (int)(e % 256);
        int s = m / 16, b = m % 16;
        hr[e] = (s == 0) ? 0.f : hs[(((size_t)(s - 1)) * 16 + b) * 256 + d];
    }
    size_t n2 = (size_t)208 * 128;
    GS_LOOP(e, n2) {
        int m = (int)(e / 128), f = (int)(e % 128);
        int s = m / 16, b = m % 16;
        zr[e] = muf[((size_t)b * 16 + s) * 128 + f];
    }
}

__global__ void k_finalize(const float* acc, float* out) {
    if (threadIdx.x == 0 && blockIdx.x == 0) {
        float rec   = acc[0] / 3145728.f;
        float kld   = acc[1] / 256.f;               // 16 kl0 rows + 15*16 scan rows
        float osm   = acc[2] / (240.f * 128.f);
        float roll  = acc[3] / (624.f * 128.f);
        float drift = acc[4] / 30720.f;
        out[0] = rec + kld;   // loss (BETA = 1)
        out[1] = rec;
        out[2] = kld;
        out[3] = kld;         // kld_raw == kld in forward value
        out[4] = osm;
        out[5] = roll;
        out[6] = drift;
    }
}

// ---------------- host orchestration ----------------
static inline dim3 gsz(size_t n) {
    size_t b = (n + 255) / 256;
    if (b > 8192) b = 8192;
    if (b < 1) b = 1;
    return dim3((unsigned)b);
}
static inline int ktof(int K) { return (K + 31) / 32; }
static inline size_t swzsz(int K, int N) { return (size_t)ktof(K) * ((N + 15) / 16) * 512; }

extern "C" void kernel_launch(void* const* d_in, const int* in_sizes, int n_in,
                              void* d_out, int out_size, void* d_ws, size_t ws_size,
                              hipStream_t stream) {
    (void)in_sizes; (void)n_in; (void)out_size; (void)ws_size;

    const float* images  = (const float*)d_in[0];
    const float* actions = (const float*)d_in[1];
    const float* eps     = (const float*)d_in[2];
    const float* We1 = (const float*)d_in[3];   const float* be1 = (const float*)d_in[4];
    const float* We2 = (const float*)d_in[5];   const float* be2 = (const float*)d_in[6];
    const float* We3 = (const float*)d_in[7];   const float* be3 = (const float*)d_in[8];
    const float* We4 = (const float*)d_in[9];   const float* be4 = (const float*)d_in[10];
    const float* We5 = (const float*)d_in[11];  const float* be5 = (const float*)d_in[12];
    const float* Wmu = (const float*)d_in[13];  const float* bmu = (const float*)d_in[14];
    const float* Wlv = (const float*)d_in[15];  const float* blv = (const float*)d_in[16];
    const float* Wd0 = (const float*)d_in[17];  const float* bd0 = (const float*)d_in[18];
    const float* Wd1 = (const float*)d_in[19];  const float* bd1 = (const float*)d_in[20];
    const float* Wd2 = (const float*)d_in[21];  const float* bd2 = (const float*)d_in[22];
    const float* Wd3 = (const float*)d_in[23];  const float* bd3 = (const float*)d_in[24];
    const float* Wd4 = (const float*)d_in[25];  const float* bd4 = (const float*)d_in[26];
    const float* Wd5 = (const float*)d_in[27];  const float* bd5 = (const float*)d_in[28];
    const float* Winp = (const float*)d_in[29]; const float* binp = (const float*)d_in[30];
    const float* Wih  = (const float*)d_in[31]; const float* bih  = (const float*)d_in[32];
    const float* Whh  = (const float*)d_in[33]; const float* bhh  = (const float*)d_in[34];
    const float* Wpri = (const float*)d_in[35]; const float* bpri = (const float*)d_in[36];

    float* out = (float*)d_out;

    // ---- workspace layout (bump allocator, 256B aligned) ----
    uint8_t* cur = (uint8_t*)d_ws;
    auto alloc  = [&](size_t bytes) -> void* {
        void* p = (void*)cur;
        cur += (bytes + 255) & ~(size_t)255;
        return p;
    };
    auto allocF = [&](size_t n) { return (float*)alloc(n * sizeof(float)); };
    auto allocH = [&](size_t n) { return (_Float16*)alloc(n * sizeof(_Float16)); };

    float* accv = allocF(16);
    float* h0   = allocF((size_t)16 * 256);
    float* hs   = allocF((size_t)15 * 16 * 256);
    float* xcat = allocF((size_t)208 * 134);
    float* xin  = allocF((size_t)208 * 256);
    float* gi   = allocF((size_t)208 * 768);
    float* gh   = allocF((size_t)208 * 768);
    float* pbuf = allocF((size_t)208 * 256);
    float* pm   = allocF((size_t)208 * 128);
    float* plv  = allocF((size_t)208 * 128);
    float* hrA  = allocF((size_t)208 * 256);
    float* hrB  = allocF((size_t)208 * 256);
    float* zr   = allocF((size_t)208 * 128);
    float* muF  = allocF((size_t)256 * 128);
    float* lvF  = allocF((size_t)256 * 128);
    float* zlat = allocF((size_t)256 * 128);
    float* feat = allocF((size_t)256 * 512);
    float* f1   = allocF((size_t)256 * 64 * 32 * 32);
    float* f2   = allocF((size_t)256 * 128 * 16 * 16);
    float* f3   = allocF((size_t)256 * 256 * 8 * 8);
    float* f4   = allocF((size_t)256 * 512 * 4 * 4);
    float* d0o  = allocF((size_t)256 * 512);
    float* d1o  = allocF((size_t)256 * 4096);
    float* d2o  = allocF((size_t)256 * 128 * 8 * 8);
    float* d3o  = allocF((size_t)256 * 64 * 16 * 16);
    float* d4o  = allocF((size_t)256 * 32 * 32 * 32);
    _Float16* wpack = allocH(swzsz(8192, 512));     // max conv pack (e5)
    _Float16* wd1h  = allocH(swzsz(512, 4096));
    _Float16* wmup  = allocH(swzsz(512, 128));
    _Float16* wlvp  = allocH(swzsz(512, 128));
    _Float16* wd0p  = allocH(swzsz(128, 512));
    _Float16* winpp = allocH(swzsz(134, 256));
    _Float16* wihp  = allocH(swzsz(256, 768));
    _Float16* whhp  = allocH(swzsz(256, 768));
    _Float16* wprip = allocH(swzsz(256, 256));
    float* patch = allocF((size_t)16384 * 4096);    // im2col chunk buffer

    const int CHUNK = 16384;

    auto gemm = [&](const float* A, const _Float16* Bm, const float* bias, float* C,
                    int M, int N, int K, int mOff, int bDiv, int act, int P) {
        dim3 grid((unsigned)((N + BN - 1) / BN), (unsigned)((M + BM - 1) / BM));
        k_wmma_gemm<<<grid, 128, 0, stream>>>(A, Bm, bias, C, M, N, K, mOff, bDiv, act, P);
    };

    auto convLayer = [&](const float* in, const float* Wf, const float* bf, float* outp,
                         int NI, int IC, int IH, int OC, int OH, int KH, int S, int Pd,
                         int act, bool transposed) {
        const int KHW = KH * KH;
        const int K = IC * KHW;
        const size_t wtot = swzsz(K, OC);
        if (!transposed)
            k_pack_conv_w<<<gsz(wtot), 256, 0, stream>>>(Wf, wpack, OC, IC, KHW, ktof(K));
        else
            k_pack_deconv_w<<<gsz(wtot), 256, 0, stream>>>(Wf, wpack, IC, OC, KHW, ktof(K));
        const int M = NI * OH * OH;
        for (int m0 = 0; m0 < M; m0 += CHUNK) {
            int mc = (M - m0 < CHUNK) ? (M - m0) : CHUNK;
            if (!transposed)
                k_im2col_conv<<<gsz((size_t)mc * K), 256, 0, stream>>>(
                    in, patch, m0, mc, IC, IH, IH, OH, OH, KH, KH, S, Pd);
            else
                k_im2col_deconv<<<gsz((size_t)mc * K), 256, 0, stream>>>(
                    in, patch, m0, mc, IC, IH, IH, OH, OH, KH, KH, S, Pd);
            gemm(patch, wpack, bf, outp, mc, OC, K, m0, 1, act, OH * OH);
        }
    };

    // ---- init ----
    k_zero<<<gsz(16), 256, 0, stream>>>(accv, 16);
    k_zero<<<gsz(4096), 256, 0, stream>>>(h0, (size_t)16 * 256);

    // ---- pack reused linear weights (swizzled) ----
    k_pack_linear_w<<<gsz(swzsz(512, 128)), 256, 0, stream>>>(Wmu, wmup, 128, 512, ktof(512));
    k_pack_linear_w<<<gsz(swzsz(512, 128)), 256, 0, stream>>>(Wlv, wlvp, 128, 512, ktof(512));
    k_pack_linear_w<<<gsz(swzsz(128, 512)), 256, 0, stream>>>(Wd0, wd0p, 512, 128, ktof(128));
    k_pack_linear_w<<<gsz(swzsz(134, 256)), 256, 0, stream>>>(Winp, winpp, 256, 134, ktof(134));
    k_pack_linear_w<<<gsz(swzsz(256, 768)), 256, 0, stream>>>(Wih, wihp, 768, 256, ktof(256));
    k_pack_linear_w<<<gsz(swzsz(256, 768)), 256, 0, stream>>>(Whh, whhp, 768, 256, ktof(256));
    k_pack_linear_w<<<gsz(swzsz(256, 256)), 256, 0, stream>>>(Wpri, wprip, 256, 256, ktof(256));
    k_pack_rowmajor_w<<<gsz(swzsz(512, 4096)), 256, 0, stream>>>(Wd1, wd1h, 512, 4096, ktof(512));

    // ---- VAE encoder (silu) ----
    convLayer(images, We1, be1, f1, 256,   3, 64,  64, 32, 4, 2, 1, 1, false);
    convLayer(f1,     We2, be2, f2, 256,  64, 32, 128, 16, 4, 2, 1, 1, false);
    convLayer(f2,     We3, be3, f3, 256, 128, 16, 256,  8, 4, 2, 1, 1, false);
    convLayer(f3,     We4, be4, f4, 256, 256,  8, 512,  4, 4, 2, 1, 1, false);
    convLayer(f4,     We5, be5, feat, 256, 512, 4, 512, 1, 4, 1, 0, 1, false);

    // ---- latent heads + reparameterize ----
    gemm(feat, wmup, bmu, muF, 256, 128, 512, 0, 1, 0, 0);
    gemm(feat, wlvp, blv, lvF, 256, 128, 512, 0, 1, 0, 0);
    k_reparam<<<gsz(32768), 256, 0, stream>>>(muF, lvF, eps, zlat, (size_t)256 * 128);

    // ---- VAE decoder ----
    gemm(zlat, wd0p, bd0, d0o, 256, 512, 128, 0, 1, 0, 0);
    // d1 deconv 1x1 -> 4x4 VALID is a pure GEMM: Wd1 IOHW layout == [K=512][N=oc*16]
    gemm(d0o, wd1h, bd1, d1o, 256, 4096, 512, 0, 16, 1, 0);
    convLayer(d1o, Wd2, bd2, d2o, 256, 256,  4, 128,  8, 4, 2, 1, 1, true);
    convLayer(d2o, Wd3, bd3, d3o, 256, 128,  8,  64, 16, 4, 2, 1, 1, true);
    convLayer(d3o, Wd4, bd4, d4o, 256,  64, 16,  32, 32, 4, 2, 1, 1, true);
    convLayer(d4o, Wd5, bd5, out + 7, 256, 32, 32, 3, 64, 4, 2, 1, 2, true); // tanh -> d_out

    // ---- reconstruction loss + simple stats ----
    k_recloss<<<gsz(3145728), 256, 0, stream>>>(out + 7, images, accv, (size_t)3145728);
    k_kl0<<<gsz(2048), 256, 0, stream>>>(muF, lvF, accv);
    k_drift<<<gsz(30720), 256, 0, stream>>>(muF, accv);

    // ---- RSSM scan (gate pass is forward-identity: z_dyn == prev mean) ----
    const float* hprev = h0;
    for (int t = 0; t < 15; ++t) {
        float* hnew = hs + (size_t)t * 16 * 256;
        k_concat_scan<<<gsz(16 * 134), 256, 0, stream>>>(muF, actions, xcat, t);
        gemm(xcat, winpp, binp, xin, 16, 256, 134, 0, 1, 3, 0);   // elu
        gemm(xin,  wihp,  bih,  gi,  16, 768, 256, 0, 1, 0, 0);
        gemm(hprev, whhp, bhh,  gh,  16, 768, 256, 0, 1, 0, 0);
        k_gru_combine<<<gsz(16 * 256), 256, 0, stream>>>(gi, gh, hprev, hnew, 16);
        gemm(hnew, wprip, bpri, pbuf, 16, 256, 256, 0, 1, 0, 0);
        k_prior_post<<<gsz(16 * 128), 256, 0, stream>>>(pbuf, pm, plv, 16);
        k_kl_err_step<<<gsz(2048), 256, 0, stream>>>(pm, plv, muF, lvF, t, accv);
        hprev = hnew;
    }

    // ---- open-loop rollout (S=13, HORIZON=3), vectorized over (s,b) ----
    k_roll_init<<<gsz(208 * 256), 256, 0, stream>>>(hs, muF, hrA, zr);
    const float* zcur = zr;
    float* hcur = hrA; float* hnxt = hrB;
    for (int k = 0; k < 3; ++k) {
        k_concat_roll<<<gsz(208 * 134), 256, 0, stream>>>(zcur, actions, xcat, k);
        gemm(xcat, winpp, binp, xin, 208, 256, 134, 0, 1, 3, 0);  // elu
        gemm(xin,  wihp,  bih,  gi,  208, 768, 256, 0, 1, 0, 0);
        gemm(hcur, whhp,  bhh,  gh,  208, 768, 256, 0, 1, 0, 0);
        k_gru_combine<<<gsz(208 * 256), 256, 0, stream>>>(gi, gh, hcur, hnxt, 208);
        gemm(hnxt, wprip, bpri, pbuf, 208, 256, 256, 0, 1, 0, 0);
        k_prior_post<<<gsz(208 * 128), 256, 0, stream>>>(pbuf, pm, plv, 208);
        k_roll_err<<<gsz(208 * 128), 256, 0, stream>>>(pm, muF, k, accv);
        zcur = pm;                 // next step's latent input (stream-ordered reuse)
        float* tmp = hcur; hcur = hnxt; hnxt = tmp;
    }

    // ---- scalars ----
    k_finalize<<<1, 1, 0, stream>>>(accv, out);
}
